// BiLSTM_8933531976027
// MI455X (gfx1250) — matrix operand order repaired and checked by reading, not compiled, not run on hardware
//
#include <hip/hip_runtime.h>
#include <hip/hip_bf16.h>

typedef unsigned short u16;
typedef unsigned int   u32;

typedef __attribute__((ext_vector_type(16))) __bf16 v16bf;
typedef __attribute__((ext_vector_type(8)))  float  v8f;
typedef __attribute__((ext_vector_type(4)))  int    vi4;

union ABfrag {
    v16bf bf;
    u16   u[16];
    uint4 q[2];
};

#define TSEQ  200
#define BATCH 1024
#define MROWS (BATCH * TSEQ)   // 204800

// ---- CDNA5 async global->LDS path (ASYNCcnt), with sync fallback ----------
#if __has_builtin(__builtin_amdgcn_global_load_async_to_lds_b128)
#define HAVE_ASYNC 1
typedef __attribute__((address_space(1))) vi4* gptr4;   // global int4*
typedef __attribute__((address_space(3))) vi4* lptr4;   // LDS int4*
#else
#define HAVE_ASYNC 0
#endif

#if HAVE_ASYNC
#if __has_builtin(__builtin_amdgcn_s_wait_asynccnt)
#define WAIT_ASYNC_N(n) __builtin_amdgcn_s_wait_asynccnt(n)
#else
#define WAIT_ASYNC_N(n) asm volatile("s_wait_asynccnt %0" :: "i"(n) : "memory")
#endif
#else
#define WAIT_ASYNC_N(n) do { } while (0)
#endif

__device__ __forceinline__ u16 f2bf(float f) {
    u32 u = __float_as_uint(f);
    u32 r = u + 0x7FFFu + ((u >> 16) & 1u);   // round-to-nearest-even
    return (u16)(r >> 16);
}

__device__ __forceinline__ u32 pack2(float a, float b) {
    return (u32)f2bf(a) | ((u32)f2bf(b) << 16);
}

__device__ __forceinline__ float sigm(float x) {
    return 0.5f * tanhf(0.5f * x) + 0.5f;     // one trans op instead of exp+rcp
}

// Stream one [64 x 128] f32 xz tile for time-step t into LDS (double buffer).
// 2048 float4s, 8 per thread; async copies tracked by ASYNCcnt.
__device__ __forceinline__ void issue_tile(const float* __restrict__ xz,
                                           float* dst, int m0, int t, int tid)
{
    #pragma unroll
    for (int k = 0; k < 8; ++k) {
        int f4  = tid + k * 256;
        int row = f4 >> 5;
        int c4  = (f4 & 31) << 2;
        const float* g = xz + ((size_t)(m0 + row) * TSEQ + t) * 512 + c4;
        float* l = dst + row * 512 + c4;
#if HAVE_ASYNC
        __builtin_amdgcn_global_load_async_to_lds_b128(
            (gptr4)(g), (lptr4)(l), 0, 0);
#else
        *(float4*)l = *(const float4*)g;
#endif
    }
}

// ---------------------------------------------------------------------------
// Input projection: xz[M,512] = IN[M,128] @ K[128,512] + bias   (both dirs)
// WG = 8 waves, output tile 128(M) x 64(N); A/B staged in LDS pre-swizzled to
// WMMA fragment layout so each lane fetches fragments as two ds_load_b128.
// ---------------------------------------------------------------------------
__global__ __launch_bounds__(256)
void proj_kernel(const float* __restrict__ inF, const float* __restrict__ inB,
                 const float* __restrict__ kF,  const float* __restrict__ kB,
                 const float* __restrict__ bF,  const float* __restrict__ bB,
                 float* __restrict__ xzF, float* __restrict__ xzB)
{
    const int dir = blockIdx.z;
    const float* __restrict__ A    = dir ? inB : inF;
    const float* __restrict__ Kmat = dir ? kB  : kF;
    const float* __restrict__ bias = dir ? bB  : bF;
    float* __restrict__ xz         = dir ? xzB : xzF;

    const int m0   = blockIdx.x * 128;
    const int n0   = blockIdx.y * 64;
    const int tid  = threadIdx.x;
    const int lane = tid & 31;
    const int w    = tid >> 5;

    __shared__ u16 aL[32 * 512];   // 8 m-tiles * 4 k-tiles A fragments (32KB)
    __shared__ u16 bL[16 * 512];   // 4 n-tiles * 4 k-tiles B fragments (16KB)

    // Fill A fragments: K-runs of 8 are contiguous in both global memory and
    // the fragment layout -> two float4 loads + one ds_store_b128 per chunk.
    for (int c = tid; c < 2048; c += 256) {
        int i8 = (c & 1) << 3;            // element base 0 or 8
        int l  = (c >> 1) & 31;
        int f  = c >> 6;                  // f = mt*4 + kk
        int mt = f >> 2, kk = f & 3;
        int Kb = kk * 32 + ((l >> 4) << 3) + ((i8 >> 3) << 4);
        const float* g = &A[(size_t)(m0 + mt * 16 + (l & 15)) * 128 + Kb];
        float4 x0 = *(const float4*)g;
        float4 x1 = *(const float4*)(g + 4);
        uint4 o;
        o.x = pack2(x0.x, x0.y); o.y = pack2(x0.z, x0.w);
        o.z = pack2(x1.x, x1.y); o.w = pack2(x1.z, x1.w);
        *(uint4*)&aL[f * 512 + l * 16 + i8] = o;
    }
    // Fill B fragments (column gather; one-time, L2-resident weights)
    for (int idx = tid; idx < 16 * 512; idx += 256) {
        int i  = idx & 15;
        int l  = (idx >> 4) & 31;
        int f  = idx >> 9;                // f = nt*4 + kk
        int nt = f >> 2, kk = f & 3;
        int Kl  = kk * 32 + ((l >> 4) << 3) + (i & 7) + ((i >> 3) << 4);
        int col = n0 + nt * 16 + (l & 15);
        bL[idx] = f2bf(Kmat[(size_t)Kl * 512 + col]);
    }
    __syncthreads();

    const int mtb  = (w >> 1) * 2;     // wave covers 2 row tiles (32 rows)
    const int ntb  = (w & 1) * 2;      // and 2 col tiles (32 cols)
    const int rowl = (lane >> 4) << 3;
    const int coll = lane & 15;

    v8f acc[2][2];
    #pragma unroll
    for (int ni = 0; ni < 2; ++ni) {
        float bv = bias[n0 + (ntb + ni) * 16 + coll];
        #pragma unroll
        for (int mi = 0; mi < 2; ++mi)
            #pragma unroll
            for (int r = 0; r < 8; ++r) acc[mi][ni][r] = bv;
    }

    #pragma unroll
    for (int kk = 0; kk < 4; ++kk) {
        ABfrag af[2], bb[2];
        #pragma unroll
        for (int mi = 0; mi < 2; ++mi) {
            const u16* p = &aL[((mtb + mi) * 4 + kk) * 512 + lane * 16];
            af[mi].q[0] = *(const uint4*)p;
            af[mi].q[1] = *(const uint4*)(p + 8);
        }
        #pragma unroll
        for (int ni = 0; ni < 2; ++ni) {
            const u16* p = &bL[((ntb + ni) * 4 + kk) * 512 + lane * 16];
            bb[ni].q[0] = *(const uint4*)p;
            bb[ni].q[1] = *(const uint4*)(p + 8);
        }
        #pragma unroll
        for (int mi = 0; mi < 2; ++mi)
            #pragma unroll
            for (int ni = 0; ni < 2; ++ni)
                acc[mi][ni] = __builtin_amdgcn_wmma_f32_16x16x32_bf16(
                    false, af[mi].bf, false, bb[ni].bf,
                    (short)0, acc[mi][ni], false, false);
    }

    #pragma unroll
    for (int mi = 0; mi < 2; ++mi)
        #pragma unroll
        for (int ni = 0; ni < 2; ++ni) {
            int col = n0 + (ntb + ni) * 16 + coll;
            size_t rbase = (size_t)(m0 + (mtb + mi) * 16 + rowl);
            #pragma unroll
            for (int r = 0; r < 8; ++r)
                xz[(rbase + r) * 512 + col] = acc[mi][ni][r];
        }
}

// ---------------------------------------------------------------------------
// LSTM recurrence. One WG = 16 batch rows for all 200 steps; grid.y selects
// direction (bw iterates t descending == reversed-input/re-reversed-output).
// Recurrent weights live in VGPRs for the whole kernel (16 bf16 B-fragments
// per wave). h is exchanged wave-to-wave via a 4KB LDS tile pre-swizzled to
// the A layout. xz gate pre-activations are double-buffered in LDS with
// async global->LDS copies so HBM latency is hidden behind each step.
// ---------------------------------------------------------------------------
__global__ __launch_bounds__(256)
void rec_kernel(const float* __restrict__ xzF, const float* __restrict__ xzB,
                const float* __restrict__ rkF, const float* __restrict__ rkB,
                float* __restrict__ hF, float* __restrict__ hB)
{
    const int dir = blockIdx.y;
    const float* __restrict__ xz = dir ? xzB : xzF;
    const float* __restrict__ rk = dir ? rkB : rkF;
    float* __restrict__ hout     = dir ? hB  : hF;

    const int m0   = blockIdx.x * 16;
    const int tid  = threadIdx.x;
    const int lane = tid & 31;
    const int w    = tid >> 5;
    const int rowl = (lane >> 4) << 3;
    const int coll = lane & 15;

    __shared__ u16   aL[4 * 512];       // h tile in A-fragment layout (4KB)
    __shared__ float zL[16 * 512];      // z = xz_t + h@rk (32KB)
    __shared__ float xb[2][16 * 512];   // xz tile double buffer (64KB)

    // Preload this wave's recurrent-weight B fragments into registers.
    ABfrag bfr[4][4];                   // [n-subtile][k-tile]
    #pragma unroll
    for (int nt4 = 0; nt4 < 4; ++nt4) {
        const int n = (w * 4 + nt4) * 16;
        #pragma unroll
        for (int kk = 0; kk < 4; ++kk) {
            #pragma unroll
            for (int i = 0; i < 16; ++i) {
                int Kl = kk * 32 + ((lane >> 4) << 3) + (i & 7) + ((i >> 3) << 4);
                bfr[nt4][kk].u[i] = f2bf(rk[(size_t)Kl * 512 + n + coll]);
            }
        }
    }

    for (int idx = tid; idx < 4 * 512; idx += 256) aL[idx] = 0;  // h0 = 0

    float c[8];
    #pragma unroll
    for (int j = 0; j < 8; ++j) c[j] = 0.f;                      // c0 = 0

    const int grow = tid >> 4;        // gate-phase row (0..15)
    const int us   = (tid & 15) * 8;  // gate-phase unit base (8 units/thread)

    issue_tile(xz, xb[0], m0, dir ? (TSEQ - 1) : 0, tid);        // prefetch t0
    __syncthreads();

    for (int s = 0; s < TSEQ; ++s) {
        const int t = dir ? (TSEQ - 1 - s) : s;

        // Prefetch next step's xz tile into the other buffer, then make the
        // current tile (8 older async ops per wave) visible.
        if (s + 1 < TSEQ) {
            issue_tile(xz, xb[(s + 1) & 1], m0,
                       dir ? (TSEQ - 2 - s) : (s + 1), tid);
            WAIT_ASYNC_N(8);
        } else {
            WAIT_ASYNC_N(0);
        }
        __syncthreads();

        // ---- z = xz[:,t,:] + h @ rk  (16 WMMAs per wave) ----
        ABfrag af[4];
        #pragma unroll
        for (int kk = 0; kk < 4; ++kk) {
            const u16* p = &aL[kk * 512 + lane * 16];
            af[kk].q[0] = *(const uint4*)p;
            af[kk].q[1] = *(const uint4*)(p + 8);
        }
        const float* xt = &xb[s & 1][0];
        #pragma unroll
        for (int nt4 = 0; nt4 < 4; ++nt4) {
            const int n = (w * 4 + nt4) * 16;
            v8f acc;
            #pragma unroll
            for (int r = 0; r < 8; ++r)
                acc[r] = xt[(rowl + r) * 512 + n + coll];   // C := xz_t
            #pragma unroll
            for (int kk = 0; kk < 4; ++kk)
                acc = __builtin_amdgcn_wmma_f32_16x16x32_bf16(
                    false, af[kk].bf, false, bfr[nt4][kk].bf,
                    (short)0, acc, false, false);
            #pragma unroll
            for (int r = 0; r < 8; ++r)
                zL[(rowl + r) * 512 + n + coll] = acc[r];
        }
        __syncthreads();

        // ---- gates: i,f,g,o = z[:,0:128|128:256|256:384|384:512] ----
        float zi[8], zf[8], zg[8], zo[8];
        {
            const float* zp = &zL[grow * 512];
            *(float4*)&zi[0] = *(const float4*)&zp[us];
            *(float4*)&zi[4] = *(const float4*)&zp[us + 4];
            *(float4*)&zf[0] = *(const float4*)&zp[128 + us];
            *(float4*)&zf[4] = *(const float4*)&zp[128 + us + 4];
            *(float4*)&zg[0] = *(const float4*)&zp[256 + us];
            *(float4*)&zg[4] = *(const float4*)&zp[256 + us + 4];
            *(float4*)&zo[0] = *(const float4*)&zp[384 + us];
            *(float4*)&zo[4] = *(const float4*)&zp[384 + us + 4];
        }
        float hv[8];
        #pragma unroll
        for (int j = 0; j < 8; ++j) {
            float ig = sigm(zi[j]);
            float fg = sigm(zf[j]);
            float gg = tanhf(zg[j]);
            float og = sigm(zo[j]);
            c[j] = fg * c[j] + ig * gg;
            float h = og * tanhf(c[j]);
            hv[j] = h;
            // scatter h (bf16) into the swizzled A-fragment tile for next step
            int uu = us + j;
            int kk = uu >> 5, koff = uu & 31;
            int lhi = (koff >> 3) & 1;
            int ii  = (koff & 7) + ((koff >> 4) << 3);
            aL[kk * 512 + (grow + 16 * lhi) * 16 + ii] = f2bf(h);
        }
        float* hp = &hout[((size_t)(m0 + grow) * TSEQ + t) * 128 + us];
        *(float4*)&hp[0] = *(const float4*)&hv[0];
        *(float4*)&hp[4] = *(const float4*)&hv[4];
        __syncthreads();
    }
}

// ---------------------------------------------------------------------------
// merge_mode='ave'
// ---------------------------------------------------------------------------
__global__ __launch_bounds__(256)
void avg_kernel(const float* __restrict__ a, const float* __restrict__ b,
                float* __restrict__ out, int n4)
{
    int i = blockIdx.x * blockDim.x + threadIdx.x;
    if (i < n4) {
        float4 x = ((const float4*)a)[i];
        float4 y = ((const float4*)b)[i];
        float4 r;
        r.x = 0.5f * (x.x + y.x);
        r.y = 0.5f * (x.y + y.y);
        r.z = 0.5f * (x.z + y.z);
        r.w = 0.5f * (x.w + y.w);
        ((float4*)out)[i] = r;
    }
}

extern "C" void kernel_launch(void* const* d_in, const int* in_sizes, int n_in,
                              void* d_out, int out_size, void* d_ws, size_t ws_size,
                              hipStream_t stream)
{
    (void)in_sizes; (void)n_in; (void)out_size; (void)ws_size;

    const float* x     = (const float*)d_in[0];
    const float* fw_k  = (const float*)d_in[1];
    const float* fw_rk = (const float*)d_in[2];
    const float* fw_b  = (const float*)d_in[3];
    const float* bw_k  = (const float*)d_in[4];
    const float* bw_rk = (const float*)d_in[5];
    const float* bw_b  = (const float*)d_in[6];
    float* out = (float*)d_out;

    const size_t xz_elems = (size_t)MROWS * 512;   // 420 MB each
    const size_t h_elems  = (size_t)MROWS * 128;   // 105 MB each
    float* xzF = (float*)d_ws;
    float* xzB = xzF + xz_elems;
    float* hF  = xzB + xz_elems;
    float* hB  = hF  + h_elems;

    dim3 pgrid(MROWS / 128, 512 / 64, 2);          // 1600 x 8 x 2
    dim3 rgrid(BATCH / 16, 2, 1);                  // 64 x 2

    for (int layer = 0; layer < 2; ++layer) {
        const float* kF = fw_k  + (size_t)layer * 128 * 512;
        const float* kB = bw_k  + (size_t)layer * 128 * 512;
        const float* rF = fw_rk + (size_t)layer * 128 * 512;
        const float* rB = bw_rk + (size_t)layer * 128 * 512;
        const float* bF = fw_b  + (size_t)layer * 512;
        const float* bB = bw_b  + (size_t)layer * 512;
        const float* inF = (layer == 0) ? x : hF;
        const float* inB = (layer == 0) ? x : hB;
        proj_kernel<<<pgrid, 256, 0, stream>>>(inF, inB, kF, kB, bF, bB, xzF, xzB);
        rec_kernel<<<rgrid, 256, 0, stream>>>(xzF, xzB, rF, rB, hF, hB);
    }
    int n4 = (int)(h_elems / 4);
    avg_kernel<<<(n4 + 255) / 256, 256, 0, stream>>>(hF, hB, out, n4);
}
